// HierarchicalPooling_4363686773150
// MI455X (gfx1250) — matrix operand order repaired and checked by reading, not compiled, not run on hardware
//
#include <hip/hip_runtime.h>
#include <hip/hip_bf16.h>

typedef __bf16 v16bf __attribute__((ext_vector_type(16)));
typedef float  v8f   __attribute__((ext_vector_type(8)));

#define WLD 136   // bf16 weight row stride in LDS (128 + 8 pad) -> conflict-free frag loads
#define HLD 132   // f32 hidden row stride in LDS (128 + 4 pad)
#define WAVES 8   // waves per block (256 threads)

__device__ __forceinline__ v8f wmma_bf16(v16bf a, v16bf b, v8f c) {
  // D = A(16x32 bf16) * B(32x16 bf16) + C(16x16 f32)
  return __builtin_amdgcn_wmma_f32_16x16x32_bf16(false, a, false, b, (short)0, c,
                                                 false, false);
}

// B fragment (32x16): lane l holds column n = l&15; half hf carries
// K=kt*32+hf*8+{0..7, 16..23} (matches A). Source is row-major w[n][k].
__device__ __forceinline__ v16bf load_b_frag(const __bf16* w, int nt, int kt,
                                             int lr, int hf) {
  const __bf16* p = w + (nt * 16 + lr) * WLD + kt * 32 + hf * 8;
  v16bf r;
#pragma unroll
  for (int e = 0; e < 16; ++e) r[e] = p[e < 8 ? e : e + 8];
  return r;
}

__device__ __forceinline__ void split16(const float* v, v16bf& hi, v16bf& lo) {
#pragma unroll
  for (int e = 0; e < 16; ++e) {
    float f  = v[e];
    __bf16 h = (__bf16)f;           // RNE truncation to bf16
    hi[e] = h;
    lo[e] = (__bf16)(f - (float)h); // residual
  }
}

__device__ __forceinline__ void atomic_add_f32(float* p, float v) {
#if defined(__HIP_PLATFORM_AMD__)
  unsafeAtomicAdd(p, v);            // native global_atomic_add_f32
#else
  atomicAdd(p, v);
#endif
}

// Fused MLP (relu(x@w1^T+b1)@w2^T+b2) + atomic scatter-add into seg_accum[assign[row]].
// n_tiles = n_rows/16 (all row counts here divide by 16).
__global__ __launch_bounds__(WAVES * 32)
void mlp_scatter_kernel(const float* __restrict__ x, int n_tiles,
                        const float* __restrict__ w1, const float* __restrict__ b1,
                        const float* __restrict__ w2, const float* __restrict__ b2,
                        const int* __restrict__ assign,
                        float* __restrict__ seg_accum) {
  extern __shared__ char smem_raw[];
  __bf16* w1h = (__bf16*)smem_raw;          // [128][WLD]
  __bf16* w1l = w1h + 128 * WLD;
  __bf16* w2h = w1l + 128 * WLD;
  __bf16* w2l = w2h + 128 * WLD;
  float*  b1s = (float*)(w2l + 128 * WLD);  // [128] staged biases
  float*  b2s = b1s + 128;
  float*  hbuf = b2s + 128;                 // WAVES * 16 * HLD f32

  // Convert weights [out=128][in=128] f32 -> bf16 hi/lo in LDS (once per block),
  // and stage biases in LDS so the inner loop never touches global scalars.
  for (int i = threadIdx.x; i < 128 * 128; i += blockDim.x) {
    int r = i >> 7, c = i & 127;
    float v = w1[i];
    __bf16 h = (__bf16)v;
    w1h[r * WLD + c] = h;
    w1l[r * WLD + c] = (__bf16)(v - (float)h);
    v = w2[i];
    h = (__bf16)v;
    w2h[r * WLD + c] = h;
    w2l[r * WLD + c] = (__bf16)(v - (float)h);
  }
  if (threadIdx.x < 128) {
    b1s[threadIdx.x] = b1[threadIdx.x];
    b2s[threadIdx.x] = b2[threadIdx.x];
  }
  __syncthreads();

  const int wave = threadIdx.x >> 5;
  const int lane = threadIdx.x & 31;
  const int hf = lane >> 4;   // lane half selects K sub-range of fragments
  const int lr = lane & 15;
  float* hrow = hbuf + wave * 16 * HLD;     // private 16x128 (padded) hidden tile

  for (int tile = blockIdx.x * WAVES + wave; tile < n_tiles;
       tile += gridDim.x * WAVES) {
    const int row0 = tile << 4;

    // ---- A fragments for layer 1 (16-bit A: K = kt*32 + hf*8 + {0..7, 16..23}) ----
    v16bf a_hi[4], a_lo[4];
    {
      const float* xp = x + (size_t)(row0 + lr) * 128;
#pragma unroll
      for (int kt = 0; kt < 4; ++kt) {
        int base = kt * 32 + hf * 8;
        float v[16];
#pragma unroll
        for (int e = 0; e < 8; ++e) v[e] = xp[base + e];
#pragma unroll
        for (int e = 0; e < 8; ++e) v[8 + e] = xp[base + 16 + e];
        split16(v, a_hi[kt], a_lo[kt]);
      }
    }

    // ---- layer 1: h = relu(x @ w1^T + b1), stored to per-wave LDS tile ----
    for (int nt = 0; nt < 8; ++nt) {
      float bv = b1s[nt * 16 + lr];
      v8f acc = {bv, bv, bv, bv, bv, bv, bv, bv};
#pragma unroll
      for (int kt = 0; kt < 4; ++kt) {
        v16bf bh = load_b_frag(w1h, nt, kt, lr, hf);
        v16bf bl = load_b_frag(w1l, nt, kt, lr, hf);
        acc = wmma_bf16(a_hi[kt], bh, acc);   // hi*hi
        acc = wmma_bf16(a_lo[kt], bh, acc);   // lo*hi
        acc = wmma_bf16(a_hi[kt], bl, acc);   // hi*lo
      }
      // D layout: VGPR i -> row i (lanes 0-15) / row i+8 (lanes 16-31), col = lr
#pragma unroll
      for (int i = 0; i < 8; ++i)
        hrow[(i + 8 * hf) * HLD + nt * 16 + lr] = fmaxf(acc[i], 0.0f);
    }

    // ---- A fragments for layer 2 from LDS hidden tile ----
    v16bf h_hi[4], h_lo[4];
    {
      const float* hp = hrow + lr * HLD;
#pragma unroll
      for (int kt = 0; kt < 4; ++kt) {
        int base = kt * 32 + hf * 8;
        float v[16];
#pragma unroll
        for (int e = 0; e < 8; ++e) v[e] = hp[base + e];
#pragma unroll
        for (int e = 0; e < 8; ++e) v[8 + e] = hp[base + 16 + e];
        split16(v, h_hi[kt], h_lo[kt]);
      }
    }

    // segment ids for the 8 rows this lane-half contributes to
    int seg[8];
#pragma unroll
    for (int i = 0; i < 8; ++i) seg[i] = assign[row0 + i + 8 * hf];

    // ---- layer 2 + scatter-add ----
    for (int nt = 0; nt < 8; ++nt) {
      float bv = b2s[nt * 16 + lr];
      v8f acc = {bv, bv, bv, bv, bv, bv, bv, bv};
#pragma unroll
      for (int kt = 0; kt < 4; ++kt) {
        v16bf bh = load_b_frag(w2h, nt, kt, lr, hf);
        v16bf bl = load_b_frag(w2l, nt, kt, lr, hf);
        acc = wmma_bf16(h_hi[kt], bh, acc);
        acc = wmma_bf16(h_lo[kt], bh, acc);
        acc = wmma_bf16(h_hi[kt], bl, acc);
      }
      int col = nt * 16 + lr;
#pragma unroll
      for (int i = 0; i < 8; ++i)
        atomic_add_f32(&seg_accum[(size_t)seg[i] * 128 + col], acc[i]);
    }
  }
}

__global__ void zero_u32(unsigned int* p, int n) {
  int i = blockIdx.x * blockDim.x + threadIdx.x;
  if (i < n) p[i] = 0u;
}

__global__ void hist_kernel(const int* __restrict__ idx, int n, int* __restrict__ cnt) {
  int i = blockIdx.x * blockDim.x + threadIdx.x;
  if (i < n) atomicAdd(&cnt[idx[i]], 1);
}

__global__ void div_mean_kernel(float* __restrict__ buf, const int* __restrict__ cnt,
                                int n_seg) {
  int i = blockIdx.x * blockDim.x + threadIdx.x;
  if (i < n_seg * 128) {
    int c = cnt[i >> 7];
    buf[i] = buf[i] / fmaxf((float)c, 1.0f);   // matches s / max(cnt, 1.0)
  }
}

extern "C" void kernel_launch(void* const* d_in, const int* in_sizes, int n_in,
                              void* d_out, int out_size, void* d_ws, size_t ws_size,
                              hipStream_t stream) {
  (void)in_sizes; (void)n_in; (void)out_size; (void)ws_size;
  const float* x  = (const float*)d_in[0];
  const int* a_bc = (const int*)d_in[1];
  const int* a_ct = (const int*)d_in[2];
  const float* w1 = (const float*)d_in[3];
  const float* b1 = (const float*)d_in[4];
  const float* w2 = (const float*)d_in[5];
  const float* b2 = (const float*)d_in[6];
  const float* w3 = (const float*)d_in[7];
  const float* b3 = (const float*)d_in[8];
  const float* w4 = (const float*)d_in[9];
  const float* b4 = (const float*)d_in[10];

  const int N_B = 1048576, N_C = 50000, N_T = 2000, D = 128;
  float* out_cable = (float*)d_out;                 // accumulate x_cable in-place
  float* out_trans = out_cable + (size_t)N_C * D;   // accumulate x_trans in-place
  int* cnt_cable = (int*)d_ws;                      // 50000 ints
  int* cnt_trans = cnt_cable + N_C;                 // 2000 ints

  const int nz = N_C * D + N_T * D;
  zero_u32<<<(nz + 255) / 256, 256, 0, stream>>>((unsigned int*)d_out, nz);
  zero_u32<<<(N_C + N_T + 255) / 256, 256, 0, stream>>>((unsigned int*)cnt_cable,
                                                        N_C + N_T);

  hist_kernel<<<(N_B + 255) / 256, 256, 0, stream>>>(a_bc, N_B, cnt_cable);
  hist_kernel<<<(N_C + 255) / 256, 256, 0, stream>>>(a_ct, N_C, cnt_trans);

  const size_t SMEM = (size_t)4 * 128 * WLD * sizeof(__bf16)    // 139264 B weights
                    + 2 * 128 * sizeof(float)                    //   1024 B biases
                    + (size_t)WAVES * 16 * HLD * sizeof(float);  //  67584 B hidden tiles

  // Stage 1: buildings -> cable accumulators (65536 tiles, grid-stride, 2/wave)
  mlp_scatter_kernel<<<4096, WAVES * 32, SMEM, stream>>>(
      x, N_B / 16, w1, b1, w2, b2, a_bc, out_cable);
  div_mean_kernel<<<(N_C * D + 255) / 256, 256, 0, stream>>>(out_cable, cnt_cable, N_C);

  // Stage 2: cables -> transformer accumulators (3125 tiles)
  const int tiles2 = N_C / 16;
  mlp_scatter_kernel<<<(tiles2 + WAVES - 1) / WAVES, WAVES * 32, SMEM, stream>>>(
      out_cable, tiles2, w3, b3, w4, b4, a_ct, out_trans);
  div_mean_kernel<<<(N_T * D + 255) / 256, 256, 0, stream>>>(out_trans, cnt_trans, N_T);
}
